// Decoder_53077205844215
// MI455X (gfx1250) — matrix operand order repaired and checked by reading, not compiled, not run on hardware
//
#include <hip/hip_runtime.h>
#include <hip/hip_bf16.h>

// ---------- vector types matching gfx1250 WMMA builtin signatures ----------
typedef __attribute__((ext_vector_type(16))) __bf16 bf16x16;
typedef __attribute__((ext_vector_type(8)))  __bf16 bf16x8;
typedef __attribute__((ext_vector_type(4)))  __bf16 bf16x4;
typedef __attribute__((ext_vector_type(8)))  float  floatx8;
typedef __attribute__((ext_vector_type(4)))  int    i32x4;

// ---------- gfx1250 async global->LDS copy (ASYNCcnt-tracked) --------------
#if defined(__has_builtin)
#if __has_builtin(__builtin_amdgcn_global_load_async_to_lds_b128)
#define HAVE_ASYNC_LDS 1
#endif
#endif
#ifndef HAVE_ASYNC_LDS
#define HAVE_ASYNC_LDS 0
#endif

typedef __attribute__((address_space(1))) i32x4 gi32x4;   // global int4
typedef __attribute__((address_space(3))) i32x4 li32x4;   // LDS int4

__device__ __forceinline__ void copy16_g2l(const __bf16* g, __bf16* l) {
#if HAVE_ASYNC_LDS
  __builtin_amdgcn_global_load_async_to_lds_b128((gi32x4*)g, (li32x4*)l, 0, 0);
#else
  *(uint4*)l = *(const uint4*)g;
#endif
}

__device__ __forceinline__ void async_wait_all() {
#if HAVE_ASYNC_LDS
#if __has_builtin(__builtin_amdgcn_s_wait_asynccnt)
  __builtin_amdgcn_s_wait_asynccnt(0);
#else
  asm volatile("s_wait_asynccnt 0x0" ::: "memory");
#endif
#endif
}

// ---------------------------------------------------------------------------
// Kernel 1: fused stochastic-ternarize + per-column scale + transpose.
//   outT[n,k] = s[n] * ((q>1)-(q<-1)), q = w[k,n] - scale*noise[k,n]
// ---------------------------------------------------------------------------
__global__ __launch_bounds__(256)
void ternarize_t(const float* __restrict__ w, const float* __restrict__ nz,
                 const float* __restrict__ s, const int* __restrict__ scale_p,
                 __bf16* __restrict__ outT, int K, int N) {
  __shared__ float tile[32][33];
  const float scale = (float)(*scale_p);
  const int tx = threadIdx.x, ty = threadIdx.y;
  const long kt = (long)blockIdx.y * 32;
  const long nt = (long)blockIdx.x * 32;
#pragma unroll
  for (int r = 0; r < 4; ++r) {
    const long k = kt + ty + r * 8;
    const long n = nt + tx;
    const float q = w[k * N + n] - scale * nz[k * N + n];
    tile[ty + r * 8][tx] = (float)((q > 1.0f) - (q < -1.0f));
  }
  __syncthreads();
#pragma unroll
  for (int r = 0; r < 4; ++r) {
    const long n = nt + ty + r * 8;
    outT[n * K + kt + tx] = (__bf16)(s[n] * tile[tx][ty + r * 8]);
  }
}

// ---------------------------------------------------------------------------
// Kernel 2: f32 -> bf16 cast for activations.
// ---------------------------------------------------------------------------
__global__ __launch_bounds__(256)
void cast_bf16(const float* __restrict__ in, __bf16* __restrict__ out, int n) {
  const int i = (blockIdx.x * 256 + threadIdx.x) * 4;
  if (i < n) {
    const float4 f = *(const float4*)&in[i];
    bf16x4 o;
    o.x = (__bf16)f.x; o.y = (__bf16)f.y; o.z = (__bf16)f.z; o.w = (__bf16)f.w;
    *(bf16x4*)&out[i] = o;
  }
}

// ---------------------------------------------------------------------------
// Kernel 3/4: bf16 WMMA GEMM  C = act(A[M,K] @ Bt[N,K]^T + bias)
//   Block tile 128(M) x 256(N), 256 threads = 8 wave32 arranged 2(M) x 4(N).
//   Wave tile 64x64 = 4x4 V_WMMA_F32_16X16X32_BF16 accumulators, K step 32.
//   Double-buffered LDS; async global->LDS staging overlapped with WMMA.
//   LDS row stride 40 elems (80B = 20 banks) -> conflict-free b128 loads.
// ---------------------------------------------------------------------------
template <int ACT, typename OUT_T>
__global__ __launch_bounds__(256)
void gemm_bias_act(const __bf16* __restrict__ A, const __bf16* __restrict__ Bt,
                   const float* __restrict__ bias, OUT_T* __restrict__ C,
                   int K, int N) {
  constexpr int BM = 128, BN = 256, BK = 32;
  constexpr int LDT = 40;                           // padded LDS row stride
  __shared__ __bf16 As[2][BM * LDT];                // 2 x 10 KB
  __shared__ __bf16 Bs[2][BN * LDT];                // 2 x 20 KB

  const int tid  = threadIdx.x;
  const int lane = tid & 31;
  const int wave = tid >> 5;
  const int wm   = wave & 1;                        // 64-row slab
  const int wn   = wave >> 1;                       // 64-col slab
  const int half = lane >> 4;
  const int l16  = lane & 15;

  const long row0 = (long)blockIdx.y * BM;
  const long col0 = (long)blockIdx.x * BN;

  floatx8 acc[4][4];
#pragma unroll
  for (int i = 0; i < 4; ++i)
#pragma unroll
    for (int j = 0; j < 4; ++j) acc[i][j] = (floatx8)0.0f;

  // ---- staging: 512 A-chunks + 1024 B-chunks of 16B, 6 per thread ----
  auto stage = [&](int buf, int kb) {
#pragma unroll
    for (int u = 0; u < 2; ++u) {
      const int c = tid * 2 + u;                    // A: 128 rows x 4 chunks
      const int r = c >> 2, ko = (c & 3) * 8;
      copy16_g2l(&A[(row0 + r) * (long)K + kb + ko], &As[buf][r * LDT + ko]);
    }
#pragma unroll
    for (int u = 0; u < 4; ++u) {
      const int c = tid * 4 + u;                    // B: 256 rows x 4 chunks
      const int r = c >> 2, ko = (c & 3) * 8;
      copy16_g2l(&Bt[(col0 + r) * (long)K + kb + ko], &Bs[buf][r * LDT + ko]);
    }
  };

  const int nsteps = K / BK;
  stage(0, 0);
  async_wait_all();
  __syncthreads();

  for (int s = 0; s < nsteps; ++s) {
    const int cur = s & 1;
    if (s + 1 < nsteps) stage(cur ^ 1, (s + 1) * BK);   // overlap with WMMA

    // ---- fragments from LDS per the 16-bit A/B WMMA VGPR layouts ----
    bf16x16 af[4], bfr[4];
#pragma unroll
    for (int i = 0; i < 4; ++i) {
      const int r = wm * 64 + i * 16 + l16;
      const bf16x8 lo = *(const bf16x8*)&As[cur][r * LDT + half * 8];
      const bf16x8 hi = *(const bf16x8*)&As[cur][r * LDT + 16 + half * 8];
      af[i] = __builtin_shufflevector(lo, hi, 0,1,2,3,4,5,6,7,8,9,10,11,12,13,14,15);
    }
#pragma unroll
    for (int j = 0; j < 4; ++j) {
      const int n = wn * 64 + j * 16 + l16;
      const bf16x8 lo = *(const bf16x8*)&Bs[cur][n * LDT + half * 8];
      const bf16x8 hi = *(const bf16x8*)&Bs[cur][n * LDT + 16 + half * 8];
      bfr[j] = __builtin_shufflevector(lo, hi, 0,1,2,3,4,5,6,7,8,9,10,11,12,13,14,15);
    }

    // ---- 16 WMMAs per wave per K step ----
#pragma unroll
    for (int i = 0; i < 4; ++i)
#pragma unroll
      for (int j = 0; j < 4; ++j)
        acc[i][j] = __builtin_amdgcn_wmma_f32_16x16x32_bf16(
            false, af[i], false, bfr[j], (short)0, acc[i][j], false, false);

    async_wait_all();      // next tile landed in LDS (this wave's copies)
    __syncthreads();       // all waves done: nxt ready, cur reusable
  }

  // ---- epilogue: bias + optional tanh, per the 16x16 f32 D layout ----
  // VGPR r: lanes 0-15 -> M=r, N=lane; lanes 16-31 -> M=r+8, N=lane-16.
#pragma unroll
  for (int i = 0; i < 4; ++i) {
#pragma unroll
    for (int j = 0; j < 4; ++j) {
      const long ncol = col0 + wn * 64 + j * 16 + l16;
      const float bv = bias[ncol];
#pragma unroll
      for (int r = 0; r < 8; ++r) {
        const long mrow = row0 + wm * 64 + i * 16 + half * 8 + r;
        float v = acc[i][j][r] + bv;
        if (ACT) v = tanhf(v);
        C[mrow * (long)N + ncol] = (OUT_T)v;
      }
    }
  }
}

// ---------------------------------------------------------------------------
// Host-side launch.
// ---------------------------------------------------------------------------
extern "C" void kernel_launch(void* const* d_in, const int* in_sizes, int n_in,
                              void* d_out, int out_size, void* d_ws, size_t ws_size,
                              hipStream_t stream) {
  const float* x  = (const float*)d_in[0];
  const float* w1 = (const float*)d_in[1];
  const float* s1 = (const float*)d_in[2];
  const float* b1 = (const float*)d_in[3];
  const float* w2 = (const float*)d_in[4];
  const float* s2 = (const float*)d_in[5];
  const float* b2 = (const float*)d_in[6];
  const float* n1 = (const float*)d_in[7];
  const float* n2 = (const float*)d_in[8];
  const int* scale = (const int*)d_in[9];

  const int B = 4096, DIN = 1024, DH = 16384, DOUT = 3072;

  // workspace layout (bytes): xb 8MB | t1T 32MB | t2T 96MB | h 128MB = 264MB
  char* ws = (char*)d_ws;
  __bf16* xb  = (__bf16*)(ws);
  __bf16* t1T = (__bf16*)(ws + ((size_t)8 << 20));
  __bf16* t2T = (__bf16*)(ws + ((size_t)40 << 20));
  __bf16* h   = (__bf16*)(ws + ((size_t)136 << 20));

  cast_bf16<<<(B * DIN / 4 + 255) / 256, 256, 0, stream>>>(x, xb, B * DIN);

  dim3 tb(32, 8);
  ternarize_t<<<dim3(DH / 32, DIN / 32), tb, 0, stream>>>(w1, n1, s1, scale, t1T, DIN, DH);
  ternarize_t<<<dim3(DOUT / 32, DH / 32), tb, 0, stream>>>(w2, n2, s2, scale, t2T, DH, DOUT);

  // h = tanh(x @ W1 + b1), stored bf16
  gemm_bias_act<1, __bf16><<<dim3(DH / 256, B / 128), 256, 0, stream>>>(
      xb, t1T, b1, h, DIN, DH);

  // out = h @ W2 + b2, f32
  gemm_bias_act<0, float><<<dim3(DOUT / 256, B / 128), 256, 0, stream>>>(
      h, t2T, b2, (float*)d_out, DH, DOUT);
}
